// TransformerBlock_27745488732221
// MI455X (gfx1250) — compile-verified, matrix operand order checked by reading, not run on hardware
//
#include <hip/hip_runtime.h>
#include <hip/hip_bf16.h>
#include <math.h>

// ---------------------------------------------------------------------------
// MI455X / gfx1250 transformer block (Enformer-style rel-pos attention).
// All heavy GEMMs via v_wmma_f32_16x16x32_bf16 (bf16 in, f32 accumulate).
// GEMM: 128x128x32 tiles, double-buffered LDS (1 barrier / K-step),
// global_prefetch on streaming operands.
// ---------------------------------------------------------------------------

typedef __bf16 bf16_t;
typedef __attribute__((ext_vector_type(16))) __bf16 v16bf;
typedef __attribute__((ext_vector_type(8)))  float  v8f;

union Frag { unsigned int u[8]; v16bf v; };

#define SEQ   1536
#define DIMD  1536
#define HEADS 8
#define DKH   64
#define DVH   192
#define NTOK  6144           // BATCH*SEQ
#define QSCALE 0.125f        // 64^-0.5

// per-dword K-pair base index inside a 16x32 bf16 WMMA operand (lane half)
__device__ __forceinline__ int kpair(int v, int half) {
  return ((v < 4) ? (v << 1) : (16 + ((v - 4) << 1))) + (half << 3);
}

__device__ __forceinline__ v8f wmma_bf16(const v16bf& a, const v16bf& b, v8f c) {
  return __builtin_amdgcn_wmma_f32_16x16x32_bf16(false, a, false, b, (short)0, c,
                                                 false, false);
}

// ---------------------------------------------------------------------------
// f32 [K][N] -> bf16 [N][K] transpose+convert (weights, done once per launch)
// ---------------------------------------------------------------------------
__global__ __launch_bounds__(256) void transpose_cvt(const float* __restrict__ src,
                                                     bf16_t* __restrict__ dst,
                                                     int K, int N) {
  int idx = blockIdx.x * 256 + threadIdx.x;
  if (idx >= K * N) return;
  int k = idx / N, n = idx - k * N;
  dst[(size_t)n * K + k] = (bf16_t)src[idx];
}

// ---------------------------------------------------------------------------
// LayerNorm: one block per row of 1536, f32 in -> bf16 out
// ---------------------------------------------------------------------------
__global__ __launch_bounds__(256) void ln_kernel(const float* __restrict__ x,
                                                 const float* __restrict__ g,
                                                 const float* __restrict__ b,
                                                 bf16_t* __restrict__ out) {
  int row = blockIdx.x;
  int tid = threadIdx.x, lane = tid & 31, wave = tid >> 5;
  const float* xr = x + (size_t)row * DIMD;
  float v[6], s = 0.f, sq = 0.f;
#pragma unroll
  for (int i = 0; i < 6; i++) { v[i] = xr[tid + i * 256]; s += v[i]; sq += v[i] * v[i]; }
#pragma unroll
  for (int off = 16; off; off >>= 1) { s += __shfl_xor(s, off); sq += __shfl_xor(sq, off); }
  __shared__ float ss[8], ssq[8];
  if (lane == 0) { ss[wave] = s; ssq[wave] = sq; }
  __syncthreads();
  s = 0.f; sq = 0.f;
#pragma unroll
  for (int w = 0; w < 8; w++) { s += ss[w]; sq += ssq[w]; }
  float mean = s * (1.0f / DIMD);
  float var  = sq * (1.0f / DIMD) - mean * mean;
  float rstd = rsqrtf(var + 1e-5f);
#pragma unroll
  for (int i = 0; i < 6; i++) {
    int col = tid + i * 256;
    out[(size_t)row * DIMD + col] = (bf16_t)((v[i] - mean) * rstd * g[col] + b[col]);
  }
}

// ---------------------------------------------------------------------------
// Positional features: rows 0..3070 per reference, row 3071 zero pad. bf16 out.
// ---------------------------------------------------------------------------
__global__ __launch_bounds__(256) void pos_kernel(bf16_t* __restrict__ pos) {
  int p = blockIdx.x * 256 + threadIdx.x;
  if (p >= 3072) return;
  bf16_t* row = pos + (size_t)p * 192;
  if (p == 3071) { for (int c = 0; c < 192; c++) row[c] = (bf16_t)0.0f; return; }
  float dist = (float)p - 1535.0f;
  float ad = fabsf(dist);
  float sgn = (dist > 0.f) ? 1.0f : ((dist < 0.f) ? -1.0f : 0.0f);
  const float max_range = log2f(1536.0f);
  float vals[96];
  float gmv[32];
  float gmax = 0.f;
#pragma unroll 4
  for (int j = 0; j < 32; j++) {
    float hl = exp2f(3.0f + (float)j * (max_range - 3.0f) * (1.0f / 31.0f));
    vals[j] = exp2f(-ad / hl);                              // exponential features
    float cw = exp2f((float)(j + 1)) - 1.0f;
    vals[32 + j] = (cw > ad) ? 1.0f : 0.0f;                 // central mask
    float mean = 48.0f + (float)j * (1536.0f - 48.0f) * (1.0f / 31.0f);
    float conc = (mean / 24.0f) * (mean / 24.0f);
    float rate = mean * (1.0f / 576.0f);
    float lu = (conc - 1.0f) * logf(fmaxf(ad, 1e-20f)) - rate * ad;
    float ln_ = lgammaf(conc) - conc * logf(rate);
    float pr = expf(lu - ln_) + 1e-8f;
    gmv[j] = pr; gmax = fmaxf(gmax, pr);
  }
  float ginv = 1.0f / gmax;
#pragma unroll 4
  for (int j = 0; j < 32; j++) vals[64 + j] = gmv[j] * ginv; // gamma pdf
  for (int c = 0; c < 96; c++) { row[c] = (bf16_t)vals[c]; row[96 + c] = (bf16_t)(sgn * vals[c]); }
}

// ---------------------------------------------------------------------------
// Generic WMMA GEMM: C[M,N] = A[M,K](bf16,row-major) x Bt[N,K](bf16)
// 128x128x32 block, 256 threads (8 waves as 2x4), double-buffered LDS,
// epilogue by mode.
// ---------------------------------------------------------------------------
enum { MODE_QCQP = 0, MODE_KHEAD = 1, MODE_VT = 2, MODE_RELK = 3,
       MODE_BIAS_RELU = 4, MODE_BIAS_RESID = 5 };

#define GBM 128
#define GBN 128
#define GBK 32
#define GLDK 40
#define GTILE (GBM * GLDK)

__global__ __launch_bounds__(256) void gemm_kernel(
    const bf16_t* __restrict__ A, const bf16_t* __restrict__ Bt,
    int M, int N, int K, int mode,
    const float* __restrict__ bias, const float* __restrict__ bias2,
    const float* __restrict__ resid,
    float* __restrict__ outF, bf16_t* __restrict__ outB, bf16_t* __restrict__ outB2) {
  __shared__ bf16_t As[2 * GTILE];
  __shared__ bf16_t Bs[2 * GTILE];
  int tid = threadIdx.x;
  int lane = tid & 31, wave = tid >> 5;
  int l16 = lane & 15, halfl = lane >> 4;
  int wm = wave >> 2, wn = wave & 3;           // 2x4 wave grid
  int bM = blockIdx.x * GBM, bN = blockIdx.y * GBN;
  int lrow = tid >> 1;                         // 0..127
  int lcol = (tid & 1) * 16;                   // 0 / 16

  const bf16_t* gA = A  + (size_t)(bM + lrow) * K + lcol;
  const bf16_t* gB = Bt + (size_t)(bN + lrow) * K + lcol;

  v8f acc[4][2];
#pragma unroll
  for (int mt = 0; mt < 4; mt++)
#pragma unroll
    for (int nt = 0; nt < 2; nt++) acc[mt][nt] = {};

  // preload tile 0
  *(uint4*)(As + lrow * GLDK + lcol) = *(const uint4*)(gA);
  *(uint4*)(Bs + lrow * GLDK + lcol) = *(const uint4*)(gB);
  __syncthreads();

  int nk = K / GBK;
  for (int kt = 0; kt < nk; kt++) {
    const bf16_t* Ac = As + (kt & 1) * GTILE;
    const bf16_t* Bc = Bs + (kt & 1) * GTILE;
    bf16_t* An = (bf16_t*)As + ((kt & 1) ^ 1) * GTILE;
    bf16_t* Bn = (bf16_t*)Bs + ((kt & 1) ^ 1) * GTILE;

    bool has_next = (kt + 1 < nk);
    uint4 av, bv;
    if (has_next) {                        // issue next-tile global loads early
      av = *(const uint4*)(gA + (size_t)(kt + 1) * GBK);
      bv = *(const uint4*)(gB + (size_t)(kt + 1) * GBK);
      if (kt + 3 < nk) {                   // stream-ahead prefetch (global_prefetch)
        __builtin_prefetch((const void*)(gA + (size_t)(kt + 3) * GBK), 0, 1);
        __builtin_prefetch((const void*)(gB + (size_t)(kt + 3) * GBK), 0, 1);
      }
    }

    Frag af[4];
#pragma unroll
    for (int mt = 0; mt < 4; mt++) {
      int m = wm * 64 + mt * 16 + l16;
#pragma unroll
      for (int v = 0; v < 8; v++)
        af[mt].u[v] = *(const unsigned int*)(Ac + m * GLDK + kpair(v, halfl));
    }
#pragma unroll
    for (int nt = 0; nt < 2; nt++) {
      Frag bf;
      int n = wn * 32 + nt * 16 + l16;
#pragma unroll
      for (int v = 0; v < 8; v++)
        bf.u[v] = *(const unsigned int*)(Bc + n * GLDK + kpair(v, halfl));
#pragma unroll
      for (int mt = 0; mt < 4; mt++)
        acc[mt][nt] = wmma_bf16(af[mt].v, bf.v, acc[mt][nt]);
    }

    if (has_next) {                        // fill the other buffer, one barrier
      *(uint4*)(An + lrow * GLDK + lcol) = av;
      *(uint4*)(Bn + lrow * GLDK + lcol) = bv;
      __syncthreads();
    }
  }

  // epilogue
#pragma unroll
  for (int mt = 0; mt < 4; mt++)
#pragma unroll
    for (int nt = 0; nt < 2; nt++)
#pragma unroll
      for (int v = 0; v < 8; v++) {
        float val = acc[mt][nt][v];
        int gm = bM + wm * 64 + mt * 16 + halfl * 8 + v;
        int gn = bN + wn * 32 + nt * 16 + l16;
        switch (mode) {
          case MODE_QCQP: {            // q*scale + biases -> [b,h,i,dk] bf16 (x2)
            int h = gn >> 6, dk = gn & 63;
            int bb = gm / SEQ, ii = gm - bb * SEQ;
            size_t base = (((size_t)(bb * HEADS + h)) * SEQ + ii) * DKH + dk;
            outB[base]  = (bf16_t)(val * QSCALE + bias[gn]);
            outB2[base] = (bf16_t)(val * QSCALE + bias2[gn]);
          } break;
          case MODE_KHEAD: {           // k -> [b,h,i,dk] bf16
            int h = gn >> 6, dk = gn & 63;
            int bb = gm / SEQ, ii = gm - bb * SEQ;
            outB[(((size_t)(bb * HEADS + h)) * SEQ + ii) * DKH + dk] = (bf16_t)val;
          } break;
          case MODE_VT: {              // v -> [b,h,dv,i] bf16 (transposed)
            int h = gn / DVH, dv = gn - h * DVH;
            int bb = gm / SEQ, ii = gm - bb * SEQ;
            outB[(((size_t)(bb * HEADS + h)) * DVH + dv) * SEQ + ii] = (bf16_t)val;
          } break;
          case MODE_RELK: {            // rel_k -> [h, p, dk] bf16
            int h = gn >> 6, dk = gn & 63;
            outB[((size_t)h * 3072 + gm) * DKH + dk] = (bf16_t)val;
          } break;
          case MODE_BIAS_RELU: {       // relu(C + b) -> bf16 row-major
            float r = val + bias[gn];
            outB[(size_t)gm * N + gn] = (bf16_t)fmaxf(r, 0.0f);
          } break;
          case MODE_BIAS_RESID: {      // C + b + resid -> f32 row-major
            outF[(size_t)gm * N + gn] = val + bias[gn] + resid[(size_t)gm * N + gn];
          } break;
        }
      }
}

// ---------------------------------------------------------------------------
// Attention: one workgroup per (b,h, 32-row query tile).
// LDS: band f32 [32][1584] (rel+content logits, shift-fused), probs bf16
// [32][1544], qc/qp staging [32][72].  ~310 KB (<= 320 KB WGP LDS).
// ---------------------------------------------------------------------------
#define AT_LDBAND 1584
#define AT_LDP    1544
#define AT_LDQ    72

__global__ __launch_bounds__(256) void attn_kernel(
    const bf16_t* __restrict__ qc, const bf16_t* __restrict__ qp,
    const bf16_t* __restrict__ kmat, const bf16_t* __restrict__ vt,
    const bf16_t* __restrict__ relk, bf16_t* __restrict__ attn_out) {
  extern __shared__ char smem[];
  float*  band  = (float*)smem;                              // 32*1584 f32
  bf16_t* probs = (bf16_t*)(band + 32 * AT_LDBAND);          // 32*1544 bf16
  bf16_t* qpS   = probs + 32 * AT_LDP;                       // 32*72
  bf16_t* qcS   = qpS + 32 * AT_LDQ;                         // 32*72

  int tid = threadIdx.x, lane = tid & 31, wave = tid >> 5;
  int l16 = lane & 15, halfl = lane >> 4;
  int bh = blockIdx.y;                 // 0..31
  int bb = bh >> 3, h = bh & 7;
  int i0 = blockIdx.x * 32;
  int p0 = 1504 - i0;                  // band start: p = n-1 + j - i, i=i0..i0+31

  { // stage qc/qp tiles (32x64 each) into LDS
    int u = tid >> 3, c = (tid & 7) * 8;
    size_t g = ((size_t)bh * SEQ + i0 + u) * DKH + c;
    *(uint4*)(qcS + u * AT_LDQ + c) = *(const uint4*)(qc + g);
    *(uint4*)(qpS + u * AT_LDQ + c) = *(const uint4*)(qp + g);
  }
  __syncthreads();

  Frag aq[2][2];
  // ---- Phase 1: Band[u,t] = (q+pb)_u . rel_k[p0+t], 98 col tiles of 16 ----
#pragma unroll
  for (int mt = 0; mt < 2; mt++)
#pragma unroll
    for (int ks = 0; ks < 2; ks++)
#pragma unroll
      for (int v = 0; v < 8; v++)
        aq[mt][ks].u[v] = *(const unsigned int*)(qpS + (mt * 16 + l16) * AT_LDQ +
                                                 ks * 32 + kpair(v, halfl));
  for (int ct = wave; ct < 98; ct += 8) {
    int t = ct * 16 + l16;
    const bf16_t* br = relk + ((size_t)h * 3072 + (p0 + t)) * DKH;
    if (ct + 8 < 98)   // prefetch this lane's row for the next iteration
      __builtin_prefetch((const void*)(br + 8 * 16 * DKH), 0, 1);
    Frag bq[2];
#pragma unroll
    for (int ks = 0; ks < 2; ks++)
#pragma unroll
      for (int v = 0; v < 8; v++)
        bq[ks].u[v] = *(const unsigned int*)(br + ks * 32 + kpair(v, halfl));
    v8f a0 = {}, a1 = {};
#pragma unroll
    for (int ks = 0; ks < 2; ks++) {
      a0 = wmma_bf16(aq[0][ks].v, bq[ks].v, a0);
      a1 = wmma_bf16(aq[1][ks].v, bq[ks].v, a1);
    }
#pragma unroll
    for (int v = 0; v < 8; v++) {
      band[(halfl * 8 + v) * AT_LDBAND + t]        = a0[v];
      band[(16 + halfl * 8 + v) * AT_LDBAND + t]   = a1[v];
    }
  }
  __syncthreads();

  // ---- Phase 2: content logits, accumulated in-place at shifted offset ----
#pragma unroll
  for (int mt = 0; mt < 2; mt++)
#pragma unroll
    for (int ks = 0; ks < 2; ks++)
#pragma unroll
      for (int v = 0; v < 8; v++)
        aq[mt][ks].u[v] = *(const unsigned int*)(qcS + (mt * 16 + l16) * AT_LDQ +
                                                 ks * 32 + kpair(v, halfl));
  for (int ct = wave; ct < 96; ct += 8) {
    int j = ct * 16 + l16;
    const bf16_t* kr = kmat + ((size_t)bh * SEQ + j) * DKH;
    if (ct + 8 < 96)
      __builtin_prefetch((const void*)(kr + 8 * 16 * DKH), 0, 1);
    Frag bq[2];
#pragma unroll
    for (int ks = 0; ks < 2; ks++)
#pragma unroll
      for (int v = 0; v < 8; v++)
        bq[ks].u[v] = *(const unsigned int*)(kr + ks * 32 + kpair(v, halfl));
    v8f a0 = {}, a1 = {};
#pragma unroll
    for (int ks = 0; ks < 2; ks++) {
      a0 = wmma_bf16(aq[0][ks].v, bq[ks].v, a0);
      a1 = wmma_bf16(aq[1][ks].v, bq[ks].v, a1);
    }
#pragma unroll
    for (int v = 0; v < 8; v++) {
      int u0 = halfl * 8 + v;
      int u1 = 16 + halfl * 8 + v;
      band[u0 * AT_LDBAND + (j - u0 + 31)] += a0[v];   // logits[u,j] = C + R(shift)
      band[u1 * AT_LDBAND + (j - u1 + 31)] += a1[v];
    }
  }
  __syncthreads();

  // ---- Phase 3: softmax per row; write bf16 probs ----
  for (int r = 0; r < 4; r++) {
    int u = wave * 4 + r;
    float* brow = band + u * AT_LDBAND + (31 - u);     // brow[j] = logits[u,j]
    float mx = -1e30f;
    for (int j = lane; j < SEQ; j += 32) mx = fmaxf(mx, brow[j]);
#pragma unroll
    for (int off = 16; off; off >>= 1) mx = fmaxf(mx, __shfl_xor(mx, off));
    float s = 0.f;
    for (int j = lane; j < SEQ; j += 32) { float e = __expf(brow[j] - mx); brow[j] = e; s += e; }
#pragma unroll
    for (int off = 16; off; off >>= 1) s += __shfl_xor(s, off);
    float inv = 1.0f / s;
    for (int j = lane; j < SEQ; j += 32) probs[u * AT_LDP + j] = (bf16_t)(brow[j] * inv);
  }
  __syncthreads();

  // ---- Phase 4: out[32,192] = probs @ V ----
  for (int nt = wave; nt < 12; nt += 8) {
    int dv = nt * 16 + l16;
    const bf16_t* vr = vt + ((size_t)bh * DVH + dv) * SEQ;
    v8f a0 = {}, a1 = {};
    for (int kt = 0; kt < 48; kt++) {
      if (kt + 4 < 48)
        __builtin_prefetch((const void*)(vr + (kt + 4) * 32), 0, 1);
      Frag f0, f1, bq;
#pragma unroll
      for (int v = 0; v < 8; v++) {
        int kp = kt * 32 + kpair(v, halfl);
        f0.u[v] = *(const unsigned int*)(probs + l16 * AT_LDP + kp);
        f1.u[v] = *(const unsigned int*)(probs + (16 + l16) * AT_LDP + kp);
        bq.u[v] = *(const unsigned int*)(vr + kp);
      }
      a0 = wmma_bf16(f0.v, bq.v, a0);
      a1 = wmma_bf16(f1.v, bq.v, a1);
    }
#pragma unroll
    for (int v = 0; v < 8; v++) {
      int u0 = halfl * 8 + v, u1 = 16 + halfl * 8 + v;
      int col = h * DVH + nt * 16 + l16;
      attn_out[(size_t)(bb * SEQ + i0 + u0) * DIMD + col] = (bf16_t)a0[v];
      attn_out[(size_t)(bb * SEQ + i0 + u1) * DIMD + col] = (bf16_t)a1[v];
    }
  }
}

// ---------------------------------------------------------------------------
// Host-side orchestration
// ---------------------------------------------------------------------------
extern "C" void kernel_launch(void* const* d_in, const int* in_sizes, int n_in,
                              void* d_out, int out_size, void* d_ws, size_t ws_size,
                              hipStream_t stream) {
  const float* x     = (const float*)d_in[0];
  const float* ln1_g = (const float*)d_in[1];
  const float* ln1_b = (const float*)d_in[2];
  const float* Wq    = (const float*)d_in[3];
  const float* Wk    = (const float*)d_in[4];
  const float* Wv    = (const float*)d_in[5];
  const float* Wo    = (const float*)d_in[6];
  const float* bo    = (const float*)d_in[7];
  const float* Wrk   = (const float*)d_in[8];
  const float* rcb   = (const float*)d_in[9];
  const float* rpb   = (const float*)d_in[10];
  const float* ln2_g = (const float*)d_in[11];
  const float* ln2_b = (const float*)d_in[12];
  const float* W1    = (const float*)d_in[13];
  const float* b1    = (const float*)d_in[14];
  const float* W2    = (const float*)d_in[15];
  const float* b2    = (const float*)d_in[16];
  float* out = (float*)d_out;

  // workspace carve-out (~197 MiB total)
  char* w = (char*)d_ws;
  auto take = [&](size_t bytes) -> void* {
    void* p = (void*)w; w += (bytes + 255) & ~(size_t)255; return p;
  };
  bf16_t* XLN  = (bf16_t*)take((size_t)NTOK * DIMD * 2);
  bf16_t* WQT  = (bf16_t*)take((size_t)512 * 1536 * 2);
  bf16_t* WKT  = (bf16_t*)take((size_t)512 * 1536 * 2);
  bf16_t* WVT  = (bf16_t*)take((size_t)1536 * 1536 * 2);
  bf16_t* WOT  = (bf16_t*)take((size_t)1536 * 1536 * 2);
  bf16_t* WRKT = (bf16_t*)take((size_t)512 * 192 * 2);
  bf16_t* W1T  = (bf16_t*)take((size_t)3072 * 1536 * 2);
  bf16_t* W2T  = (bf16_t*)take((size_t)1536 * 3072 * 2);
  bf16_t* POS  = (bf16_t*)take((size_t)3072 * 192 * 2);
  bf16_t* QC   = (bf16_t*)take((size_t)32 * 1536 * 64 * 2);
  bf16_t* QP   = (bf16_t*)take((size_t)32 * 1536 * 64 * 2);
  bf16_t* KM   = (bf16_t*)take((size_t)32 * 1536 * 64 * 2);
  bf16_t* VT   = (bf16_t*)take((size_t)32 * 192 * 1536 * 2);
  bf16_t* RELK = (bf16_t*)take((size_t)8 * 3072 * 64 * 2);
  bf16_t* AO   = (bf16_t*)take((size_t)NTOK * DIMD * 2);
  float*  X1   = (float*) take((size_t)NTOK * DIMD * 4);
  bf16_t* Y2   = (bf16_t*)take((size_t)NTOK * DIMD * 2);
  bf16_t* H1   = (bf16_t*)take((size_t)NTOK * 3072 * 2);

  // 1. weight transpose+convert to bf16 [N][K]
  auto tc = [&](const float* s, bf16_t* d, int K, int N) {
    transpose_cvt<<<(K * N + 255) / 256, 256, 0, stream>>>(s, d, K, N);
  };
  tc(Wq, WQT, 1536, 512);
  tc(Wk, WKT, 1536, 512);
  tc(Wv, WVT, 1536, 1536);
  tc(Wo, WOT, 1536, 1536);
  tc(Wrk, WRKT, 192, 512);
  tc(W1, W1T, 1536, 3072);
  tc(W2, W2T, 3072, 1536);

  // 2. LN1 and positional features
  ln_kernel<<<NTOK, 256, 0, stream>>>(x, ln1_g, ln1_b, XLN);
  pos_kernel<<<12, 256, 0, stream>>>(POS);

  // 3. projections
  gemm_kernel<<<dim3(48, 4), 256, 0, stream>>>(XLN, WQT, NTOK, 512, 1536,
      MODE_QCQP, rcb, rpb, nullptr, nullptr, QC, QP);
  gemm_kernel<<<dim3(48, 4), 256, 0, stream>>>(XLN, WKT, NTOK, 512, 1536,
      MODE_KHEAD, nullptr, nullptr, nullptr, nullptr, KM, nullptr);
  gemm_kernel<<<dim3(48, 12), 256, 0, stream>>>(XLN, WVT, NTOK, 1536, 1536,
      MODE_VT, nullptr, nullptr, nullptr, nullptr, VT, nullptr);
  gemm_kernel<<<dim3(24, 4), 256, 0, stream>>>(POS, WRKT, 3072, 512, 192,
      MODE_RELK, nullptr, nullptr, nullptr, nullptr, RELK, nullptr);

  // 4. fused rel-pos attention (1536 workgroups, 310 KB LDS each)
  size_t smem = 32 * AT_LDBAND * sizeof(float) + 32 * AT_LDP * sizeof(bf16_t) +
                2 * 32 * AT_LDQ * sizeof(bf16_t);
  attn_kernel<<<dim3(48, 32), 256, smem, stream>>>(QC, QP, KM, VT, RELK, AO);

  // 5. output projection + residual -> x1 (f32)
  gemm_kernel<<<dim3(48, 12), 256, 0, stream>>>(AO, WOT, NTOK, 1536, 1536,
      MODE_BIAS_RESID, bo, nullptr, x, X1, nullptr, nullptr);

  // 6. MLP
  ln_kernel<<<NTOK, 256, 0, stream>>>(X1, ln2_g, ln2_b, Y2);
  gemm_kernel<<<dim3(48, 24), 256, 0, stream>>>(Y2, W1T, NTOK, 3072, 1536,
      MODE_BIAS_RELU, b1, nullptr, nullptr, nullptr, H1, nullptr);
  gemm_kernel<<<dim3(48, 12), 256, 0, stream>>>(H1, W2T, NTOK, 1536, 3072,
      MODE_BIAS_RESID, b2, nullptr, X1, out, nullptr, nullptr);
}
